// LocalSelfAttention_23081154249297
// MI455X (gfx1250) — matrix operand order repaired and checked by reading, not compiled, not run on hardware
//
#include <hip/hip_runtime.h>

// ---------------- problem constants (match reference) ----------------
#define BATCH 2
#define SEQ   8192
#define HIDN  1024
#define NH    16
#define DH    64
#define CHUNKSZ 64
#define NC    (SEQ / CHUNKSZ)      // 128 chunks
#define WIN   (2 * CHUNKSZ)        // 128 window (prev + current chunk)
#define ROWS  (BATCH * SEQ)        // 16384
#define NQKV  3072                 // packed Q|K|V output columns
#define QKC   2048                 // Q|K columns kept row-major

// ---------------- types ----------------
typedef __attribute__((ext_vector_type(16))) __bf16 v16bf;
typedef __attribute__((ext_vector_type(8)))  __bf16 v8bf;
typedef __attribute__((ext_vector_type(8)))  float  v8f;
typedef __attribute__((ext_vector_type(4)))  float  f32x4;
typedef __attribute__((ext_vector_type(8)))  unsigned short u16x8;
typedef __attribute__((ext_vector_type(4)))  unsigned short u16x4;
typedef __attribute__((ext_vector_type(4)))  int    i32x4;

// ---------------- gfx1250 async global->LDS support ----------------
#if defined(__has_builtin)
#if __has_builtin(__builtin_amdgcn_global_load_async_to_lds_b128)
#define USE_ASYNC_LDS 1
#endif
#endif

typedef __attribute__((address_space(1))) i32x4 gi32x4;   // global int4
typedef __attribute__((address_space(3))) i32x4 li32x4;   // LDS int4

#if defined(USE_ASYNC_LDS)
#if __has_builtin(__builtin_amdgcn_s_wait_asynccnt)
#define WAIT_ASYNC(n) __builtin_amdgcn_s_wait_asynccnt(n)
#else
#define WAIT_ASYNC(n) asm volatile("s_wait_asynccnt %0" ::"i"(n) : "memory")
#endif
#else
#define WAIT_ASYNC(n)
#endif

// float -> bf16 bits, round to nearest even
static __device__ __forceinline__ unsigned short f2us(float f) {
    unsigned u = __builtin_bit_cast(unsigned, f);
    u += 0x7FFFu + ((u >> 16) & 1u);
    return (unsigned short)(u >> 16);
}

static __device__ __forceinline__ v16bf cat8(v8bf lo, v8bf hi) {
    return __builtin_shufflevector(lo, hi, 0,1,2,3,4,5,6,7,8,9,10,11,12,13,14,15);
}
// B-fragment: 16 contiguous bf16 (K = half*16 + 0..15, one column per lane)
static __device__ __forceinline__ v16bf ldB16(const unsigned short* p) {
    return cat8(*(const v8bf*)p, *(const v8bf*)(p + 8));
}
// A-fragment: 16 contiguous bf16 (K = half*16 + 0..15)
static __device__ __forceinline__ v16bf ldA16(const unsigned short* p) {
    return cat8(*(const v8bf*)p, *(const v8bf*)(p + 8));
}
static __device__ __forceinline__ v8f wmma_bf16(v16bf a, v16bf b, v8f c) {
    return __builtin_amdgcn_wmma_f32_16x16x32_bf16(false, a, false, b,
                                                   (short)0, c, false, false);
}

// ---------------- kernel 1: X fp32 -> bf16 ----------------
__global__ void k_convert_x(const float* __restrict__ X, unsigned short* __restrict__ Xb) {
    size_t i = ((size_t)blockIdx.x * blockDim.x + threadIdx.x) * 4;
    f32x4 v = *(const f32x4*)(X + i);
    u16x4 o;
    o[0] = f2us(v[0]); o[1] = f2us(v[1]); o[2] = f2us(v[2]); o[3] = f2us(v[3]);
    *(u16x4*)(Xb + i) = o;
}

// ---------------- kernel 2: pack Wq | Wk*0.125 | Wv, K-major bf16 ----------------
__global__ void k_pack_wt(const float* __restrict__ Wq, const float* __restrict__ Wk,
                          const float* __restrict__ Wv, unsigned short* __restrict__ Wt) {
    size_t tid = (size_t)blockIdx.x * blockDim.x + threadIdx.x;   // 0 .. 3072*1024-1
    int c = (int)(tid >> 10);          // output column 0..3071
    int k = (int)(tid & 1023);         // reduction index
    int which = c >> 10;               // 0=Q 1=K 2=V
    int cc = c & 1023;
    const float* W = (which == 0) ? Wq : (which == 1 ? Wk : Wv);
    float v = W[(size_t)k * (NH * DH) + cc];
    if (which == 1) v *= 0.125f;       // fold 1/sqrt(DH) into K
    Wt[(size_t)c * HIDN + k] = f2us(v);
}

// ---------------- kernel 3: fused QKV GEMM ----------------
// C[16384,3072] = Xb[16384,1024] @ Wt^T ; cols<2048 -> QK row-major, cols>=2048 -> Vt[col][row]
// Block: 256 thr = 8 waves; block tile 64(M) x 256(N); wave tile 64 x 32.
// A tile (64x32, shared by all 8 waves) is double-buffered in LDS via async global->LDS.
__global__ void k_gemm_qkv(const unsigned short* __restrict__ Xb,
                           const unsigned short* __restrict__ Wt,
                           unsigned short* __restrict__ QK,
                           unsigned short* __restrict__ Vt) {
    __shared__ __align__(16) unsigned short ldsA[2][64][32];   // 8 KB, double buffered

    const int wave = threadIdx.x >> 5;
    const int lane = threadIdx.x & 31;
    const int lm   = lane & 15;
    const int half = lane >> 4;
    const int Moff = blockIdx.x * 64;
    const int Noff = blockIdx.y * 256 + wave * 32;

    // cooperative A-tile fill: each of 256 threads moves one 16-byte segment
    const int frow = threadIdx.x >> 2;        // 0..63
    const int fseg = threadIdx.x & 3;         // 0..3 (16B each -> 64B row)
    auto fillA = [&](int buf, int k0) {
        const unsigned short* src = Xb + (size_t)(Moff + frow) * HIDN + k0 + fseg * 8;
        unsigned short* dst = &ldsA[buf][frow][fseg * 8];
#if defined(USE_ASYNC_LDS)
        __builtin_amdgcn_global_load_async_to_lds_b128((gi32x4*)src, (li32x4*)dst, 0, 0);
#else
        *(u16x8*)dst = *(const u16x8*)src;
#endif
    };

    v8f zero = {};
    v8f acc[4][2];
#pragma unroll
    for (int mt = 0; mt < 4; ++mt)
#pragma unroll
        for (int nt = 0; nt < 2; ++nt) acc[mt][nt] = zero;

    const unsigned short* brow[2];
#pragma unroll
    for (int nt = 0; nt < 2; ++nt)
        brow[nt] = Wt + (size_t)(Noff + nt * 16 + lm) * HIDN + half * 16;

    int buf = 0;
    fillA(buf, 0);                                     // prologue fill
    for (int k0 = 0; k0 < HIDN; k0 += 32) {
        if (k0 + 32 < HIDN) {
            fillA(buf ^ 1, k0 + 32);                   // overlap next tile fill
            WAIT_ASYNC(1);                             // my fill(buf) landed
        } else {
            WAIT_ASYNC(0);
        }
        __syncthreads();                               // everyone's fill(buf) landed

        v16bf a[4], b[2];
#pragma unroll
        for (int mt = 0; mt < 4; ++mt)
            a[mt] = ldA16(&ldsA[buf][mt * 16 + lm][half * 16]);   // ds_load_b128 x2
#pragma unroll
        for (int nt = 0; nt < 2; ++nt) {
            b[nt] = ldB16(brow[nt] + k0);
            __builtin_prefetch(brow[nt] + k0 + 64, 0, 0);        // global_prefetch_b8
        }
#pragma unroll
        for (int mt = 0; mt < 4; ++mt)
#pragma unroll
            for (int nt = 0; nt < 2; ++nt)
                acc[mt][nt] = wmma_bf16(a[mt], b[nt], acc[mt][nt]);

        __syncthreads();                               // tile consumed before refill
        buf ^= 1;
    }

#pragma unroll
    for (int nt = 0; nt < 2; ++nt) {
        const int colBase = Noff + nt * 16;            // wave-uniform region select
        if (colBase < QKC) {
            const int col = colBase + lm;
#pragma unroll
            for (int mt = 0; mt < 4; ++mt) {
                const int row0 = Moff + mt * 16 + half * 8;
#pragma unroll
                for (int r = 0; r < 8; ++r)
                    QK[(size_t)(row0 + r) * QKC + col] = f2us(acc[mt][nt][r]);
            }
        } else {
            const int vcol = colBase - QKC + lm;
#pragma unroll
            for (int mt = 0; mt < 4; ++mt) {
                const int row0 = Moff + mt * 16 + half * 8;
                u16x8 o;
#pragma unroll
                for (int r = 0; r < 8; ++r) o[r] = f2us(acc[mt][nt][r]);
                *(u16x8*)(Vt + (size_t)vcol * ROWS + row0) = o;
            }
        }
    }
}

// ---------------- kernel 4: chunked local attention ----------------
// grid = (NC, NH, B), block = 128 (4 waves, wave w owns query rows 16w..16w+15)
__global__ void k_attn(const unsigned short* __restrict__ QK,
                       const unsigned short* __restrict__ Vt,
                       float* __restrict__ out) {
    __shared__ __align__(16) unsigned short pl[4][16][WIN];   // probs transpose staging

    const int chunk = blockIdx.x;
    const int h     = blockIdx.y;
    const int bz    = blockIdx.z;
    const int wave  = threadIdx.x >> 5;
    const int lane  = threadIdx.x & 31;
    const int lm    = lane & 15;
    const int half  = lane >> 4;
    const int pc    = (chunk + NC - 1) & (NC - 1);            // previous chunk (wraps)

    // ---- Q fragments: 16 query rows for this wave, dh = 64 -> two K-steps
    const unsigned short* qbase =
        QK + (size_t)(bz * SEQ + chunk * CHUNKSZ + wave * 16 + lm) * QKC + h * DH + half * 16;
    v16bf aq[2];
#pragma unroll
    for (int kt = 0; kt < 2; ++kt) aq[kt] = ldA16(qbase + kt * 32);

    // ---- dots = Q @ K^T  (16 x 128), 8 N-tiles of 16 keys
    v8f zero = {};
    v8f sacc[8];
#pragma unroll
    for (int nt = 0; nt < 8; ++nt) sacc[nt] = zero;

#pragma unroll
    for (int nt = 0; nt < 8; ++nt) {
        const int sk = (nt < 4) ? (pc * CHUNKSZ + nt * 16 + lm)
                                : (chunk * CHUNKSZ + (nt - 4) * 16 + lm);
        const unsigned short* kptr =
            QK + (size_t)(bz * SEQ + sk) * QKC + HIDN + h * DH + half * 16;
#pragma unroll
        for (int kt = 0; kt < 2; ++kt)
            sacc[nt] = wmma_bf16(aq[kt], ldB16(kptr + kt * 32), sacc[nt]);
    }

    // ---- causal mask (rolled kv indices; wrapped chunk-0 window is fully future)
    const int qrow0 = chunk * CHUNKSZ + wave * 16 + half * 8;
#pragma unroll
    for (int nt = 0; nt < 8; ++nt) {
        const int kv = (nt < 4) ? (pc * CHUNKSZ + nt * 16 + lm)
                                : (chunk * CHUNKSZ + (nt - 4) * 16 + lm);
#pragma unroll
        for (int r = 0; r < 8; ++r)
            if (qrow0 + r < kv) sacc[nt][r] -= 1e9f;
    }

    // ---- softmax per row; a row's 16 columns live in one 16-lane half
#pragma unroll
    for (int r = 0; r < 8; ++r) {
        float mx = sacc[0][r];
#pragma unroll
        for (int nt = 1; nt < 8; ++nt) mx = fmaxf(mx, sacc[nt][r]);
#pragma unroll
        for (int off = 8; off >= 1; off >>= 1)
            mx = fmaxf(mx, __shfl_xor(mx, off, 32));
        float sum = 0.f;
#pragma unroll
        for (int nt = 0; nt < 8; ++nt) {
            float e = __expf(sacc[nt][r] - mx);
            sacc[nt][r] = e;
            sum += e;
        }
#pragma unroll
        for (int off = 8; off >= 1; off >>= 1)
            sum += __shfl_xor(sum, off, 32);
        const float inv = 1.0f / sum;
#pragma unroll
        for (int nt = 0; nt < 8; ++nt)
            pl[wave][half * 8 + r][nt * 16 + lm] = f2us(sacc[nt][r] * inv);
    }
    __syncthreads();

    // ---- out = P @ V  (16 x 64), K = 128 window, 4 dh tiles
    v8f oacc[4];
#pragma unroll
    for (int dt = 0; dt < 4; ++dt) oacc[dt] = zero;

#pragma unroll
    for (int kt = 0; kt < 4; ++kt) {
        v16bf ap = ldA16(&pl[wave][lm][kt * 32 + half * 16]);
        const int wrow = kt * 32 + half * 16;
        const int sv = (wrow < CHUNKSZ) ? (pc * CHUNKSZ + wrow)
                                        : (chunk * CHUNKSZ + wrow - CHUNKSZ);
#pragma unroll
        for (int dt = 0; dt < 4; ++dt) {
            const unsigned short* vptr =
                Vt + (size_t)(h * DH + dt * 16 + lm) * ROWS + bz * SEQ + sv;
            oacc[dt] = wmma_bf16(ap, ldB16(vptr), oacc[dt]);
        }
    }

    // ---- store fp32 output [b, s, h*64 + d]
#pragma unroll
    for (int dt = 0; dt < 4; ++dt) {
        const int dcol = h * DH + dt * 16 + lm;
#pragma unroll
        for (int r = 0; r < 8; ++r) {
            const int sq = chunk * CHUNKSZ + wave * 16 + half * 8 + r;
            out[(size_t)(bz * SEQ + sq) * (NH * DH) + dcol] = oacc[dt][r];
        }
    }
}

// ---------------- host launcher ----------------
extern "C" void kernel_launch(void* const* d_in, const int* in_sizes, int n_in,
                              void* d_out, int out_size, void* d_ws, size_t ws_size,
                              hipStream_t stream) {
    (void)in_sizes; (void)n_in; (void)out_size; (void)ws_size;
    const float* X  = (const float*)d_in[0];
    const float* Wq = (const float*)d_in[1];
    const float* Wk = (const float*)d_in[2];
    const float* Wv = (const float*)d_in[3];
    float* out = (float*)d_out;

    char* ws = (char*)d_ws;
    unsigned short* Xb = (unsigned short*)(ws);                          // 32 MB
    unsigned short* Wt = (unsigned short*)(ws + 33554432);               //  6 MB
    unsigned short* QK = (unsigned short*)(ws + 39845888);               // 64 MB
    unsigned short* Vt = (unsigned short*)(ws + 106954752);              // 32 MB

    k_convert_x<<<(ROWS * HIDN) / 4 / 256, 256, 0, stream>>>(X, Xb);
    k_pack_wt<<<(NQKV * HIDN) / 256, 256, 0, stream>>>(Wq, Wk, Wv, Wt);
    k_gemm_qkv<<<dim3(ROWS / 64, NQKV / 256), 256, 0, stream>>>(Xb, Wt, QK, Vt);
    k_attn<<<dim3(NC, NH, BATCH), 128, 0, stream>>>(QK, Vt, out);
}
